// FDAttention_20401094656323
// MI455X (gfx1250) — compile-verified, matrix operand order checked by reading, not run on hardware
//
#include <hip/hip_runtime.h>
#include <hip/hip_bf16.h>
#include <cmath>

// ---------------------------------------------------------------------------
// FD attention, B=4, D=64, F=64 (DF=4096), T=2000 (padded to Tp=2048).
//   q  = (e + PE)/64       (B,Tp,DF)  bf16   (scale folded in)
//   S  = q @ k             (B,Tp,Tp)  f32    (64MB, fits in 192MB L2)
//   P  = softmax(S) masked (B,Tp,Tp)  bf16
//   out= (P @ v)^T         (B,DF,T)   f32
// WMMA: v_wmma_f32_16x16x32_bf16.
// GEMM core: workgroup 256 thr (8 waves), block tile 128x128, wave tile 32x64
// (2x4 WMMA tiles), K-step 32, double-buffered LDS, 1 barrier/iteration,
// global_prefetch_b8 at 4-tile distance.
// Workspace (288 MB): qb 64 | xb 64 | vb 64 | sc 64 | pb 32
// ---------------------------------------------------------------------------

typedef __attribute__((ext_vector_type(16))) __bf16 v16bf;
typedef __attribute__((ext_vector_type(8)))  __bf16 v8bf;
typedef __attribute__((ext_vector_type(8)))  float  v8f;

#define NB   4                 // batch
#define TT   2000              // true T
#define TP   2048              // padded T
#define DF   4096              // D*F
#define LDK  40                // padded LDS row stride (bf16): 32 + 8

// ---------------- prep kernels --------------------------------------------

__global__ __launch_bounds__(256)
void prep_q_kernel(const float* __restrict__ e, __bf16* __restrict__ qb)
{
    long idx = (long)blockIdx.x * 256 + threadIdx.x;     // over NB*TP*DF
    long df  = idx % DF;
    long t   = (idx / DF) % TP;
    long b   = idx / ((long)DF * TP);
    float val = 0.0f;
    if (t < TT) {
        // interleaved sinusoidal PE: even dims sin, odd dims cos
        float ang = (float)t * __expf(-(float)(df & ~1L) * (9.210340371976184f / (float)DF));
        float pe  = (df & 1) ? __cosf(ang) : __sinf(ang);
        val = (e[b * DF + df] + pe) * 0.015625f;         // fold 1/sqrt(4096)
    }
    qb[idx] = (__bf16)val;
}

__global__ __launch_bounds__(256)
void prep_kv_kernel(const float* __restrict__ x,
                    __bf16* __restrict__ xb,   // (NB, DF, TP)
                    __bf16* __restrict__ vb)   // (NB, TP, DF)
{
    long idx = (long)blockIdx.x * 256 + threadIdx.x;     // over NB*DF*TP
    long t   = idx % TP;
    long df  = (idx / TP) % DF;
    long b   = idx / ((long)TP * DF);
    float val = (t < TT) ? x[(b * DF + df) * (long)TT + t] : 0.0f;
    __bf16 bv = (__bf16)val;
    xb[(b * DF + df) * (long)TP + t] = bv;
    vb[(b * TP + t)  * (long)DF + df] = bv;
}

// ---------------- fragment helper -----------------------------------------

__device__ __forceinline__ v16bf load_frag(const __bf16* lds_row, int kb)
{
    v8bf lo = *(const v8bf*)(lds_row + kb);
    v8bf hi = *(const v8bf*)(lds_row + kb + 16);
    return __builtin_shufflevector(lo, hi, 0,1,2,3,4,5,6,7,8,9,10,11,12,13,14,15);
}

// ---------------- shared double-buffered WMMA GEMM core --------------------

__device__ __forceinline__ void gemm_core(const __bf16* __restrict__ Ag, long lda,
                                          const __bf16* __restrict__ Bg, long ldb,
                                          int nk, v8f (&acc)[2][4])
{
    __shared__ __bf16 As[2][128 * LDK];
    __shared__ __bf16 Bs[2][128 * LDK];

    const int tid  = threadIdx.x;
    const int lane = tid & 31;
    const int wave = tid >> 5;
    const int wr   = wave >> 1;            // 0..3 : 32-row strip
    const int wc   = wave & 1;             // 0..1 : 64-col strip
    const int l15  = lane & 15;
    const int kb   = (lane & 16) ? 8 : 0;  // K-base per 16-bit WMMA layout

    // two 8-bf16 chunks per matrix per thread: 512 chunks cover 128x32
    const int c0 = tid, c1 = tid + 256;
    const int ar0 = c0 >> 2, ac0 = (c0 & 3) * 8;
    const int ar1 = c1 >> 2, ac1 = (c1 & 3) * 8;

    // prologue: tile 0 -> buffer 0
    v8bf a0 = *(const v8bf*)(Ag + (long)ar0 * lda + ac0);
    v8bf a1 = *(const v8bf*)(Ag + (long)ar1 * lda + ac1);
    v8bf b0 = *(const v8bf*)(Bg + (long)ar0 * ldb + ac0);
    v8bf b1 = *(const v8bf*)(Bg + (long)ar1 * ldb + ac1);
    *(v8bf*)(&As[0][ar0 * LDK + ac0]) = a0;
    *(v8bf*)(&As[0][ar1 * LDK + ac1]) = a1;
    *(v8bf*)(&Bs[0][ar0 * LDK + ac0]) = b0;
    *(v8bf*)(&Bs[0][ar1 * LDK + ac1]) = b1;
    __syncthreads();

    for (int kt = 0; kt < nk; ++kt) {
        const int  cur  = kt & 1;
        const bool more = (kt + 1) < nk;
        const long koff = (long)(kt + 1) * 32;

        if (more) {
            a0 = *(const v8bf*)(Ag + (long)ar0 * lda + koff + ac0);
            a1 = *(const v8bf*)(Ag + (long)ar1 * lda + koff + ac1);
            b0 = *(const v8bf*)(Bg + (long)ar0 * ldb + koff + ac0);
            b1 = *(const v8bf*)(Bg + (long)ar1 * ldb + koff + ac1);
            if (kt + 4 < nk) {   // L2 prefetch ahead of the b128 stream
                __builtin_prefetch(Ag + (long)ar0 * lda + (long)(kt + 4) * 32 + ac0, 0, 1);
                __builtin_prefetch(Bg + (long)ar0 * ldb + (long)(kt + 4) * 32 + ac0, 0, 1);
            }
        }

        v16bf af[2], bf[4];
        #pragma unroll
        for (int mt = 0; mt < 2; ++mt)
            af[mt] = load_frag(&As[cur][(wr * 32 + mt * 16 + l15) * LDK], kb);
        #pragma unroll
        for (int nt = 0; nt < 4; ++nt)
            bf[nt] = load_frag(&Bs[cur][(wc * 64 + nt * 16 + l15) * LDK], kb);

        #pragma unroll
        for (int mt = 0; mt < 2; ++mt)
            #pragma unroll
            for (int nt = 0; nt < 4; ++nt)
                acc[mt][nt] = __builtin_amdgcn_wmma_f32_16x16x32_bf16(
                    false, af[mt], false, bf[nt], (short)0, acc[mt][nt],
                    false, false);

        if (more) {
            *(v8bf*)(&As[cur ^ 1][ar0 * LDK + ac0]) = a0;
            *(v8bf*)(&As[cur ^ 1][ar1 * LDK + ac1]) = a1;
            *(v8bf*)(&Bs[cur ^ 1][ar0 * LDK + ac0]) = b0;
            *(v8bf*)(&Bs[cur ^ 1][ar1 * LDK + ac1]) = b1;
        }
        __syncthreads();
    }
}

// ---------------- GEMM 1: S = q @ k ----------------------------------------

__global__ __launch_bounds__(256)
void gemm_qk_kernel(const __bf16* __restrict__ qb,  // (NB, TP, DF)
                    const __bf16* __restrict__ vb,  // (NB, TP, DF) = k^T rows
                    float* __restrict__ sc)         // (NB, TP, TP)
{
    const int  lane = threadIdx.x & 31;
    const int  wave = threadIdx.x >> 5;
    const int  wr   = wave >> 1;
    const int  wc   = wave & 1;
    const int  l15  = lane & 15;
    const long b    = blockIdx.z;
    const long m0   = (long)blockIdx.y * 128;
    const long n0   = (long)blockIdx.x * 128;

    v8f acc[2][4] = {};
    gemm_core(qb + (b * TP + m0) * (long)DF, DF,
              vb + (b * TP + n0) * (long)DF, DF,
              DF / 32, acc);

    #pragma unroll
    for (int mt = 0; mt < 2; ++mt)
        #pragma unroll
        for (int nt = 0; nt < 4; ++nt)
            #pragma unroll
            for (int r = 0; r < 8; ++r) {
                long row = m0 + wr * 32 + mt * 16 + r + ((lane & 16) ? 8 : 0);
                long col = n0 + wc * 64 + nt * 16 + l15;
                sc[(b * TP + row) * (long)TP + col] = acc[mt][nt][r];
            }
}

// ---------------- softmax over each row, masked at j>=TT -------------------

__global__ __launch_bounds__(256)
void softmax_kernel(const float* __restrict__ sc,  // (NB, TP, TP)
                    __bf16* __restrict__ pb)       // (NB, TP, TP)
{
    __shared__ float red[256];
    const int  tid = threadIdx.x;
    const long row = blockIdx.x;
    const long b   = blockIdx.y;
    const float* srow = sc + (b * TP + row) * (long)TP;
    __bf16*      prow = pb + (b * TP + row) * (long)TP;

    float sv[8];
    float mx = -3.0e38f;
    #pragma unroll
    for (int i = 0; i < 8; ++i) {
        int j = tid + 256 * i;
        sv[i] = srow[j];
        if (j < TT) mx = fmaxf(mx, sv[i]);
    }
    red[tid] = mx; __syncthreads();
    for (int s = 128; s > 0; s >>= 1) {
        if (tid < s) red[tid] = fmaxf(red[tid], red[tid + s]);
        __syncthreads();
    }
    mx = red[0]; __syncthreads();

    float pv[8], sum = 0.0f;
    #pragma unroll
    for (int i = 0; i < 8; ++i) {
        int j = tid + 256 * i;
        pv[i] = (j < TT) ? __expf(sv[i] - mx) : 0.0f;
        sum += pv[i];
    }
    red[tid] = sum; __syncthreads();
    for (int s = 128; s > 0; s >>= 1) {
        if (tid < s) red[tid] += red[tid + s];
        __syncthreads();
    }
    float inv = 1.0f / red[0];
    #pragma unroll
    for (int i = 0; i < 8; ++i)
        prow[tid + 256 * i] = (__bf16)(pv[i] * inv);
}

// ---------------- GEMM 2: out^T = P @ v ------------------------------------

__global__ __launch_bounds__(256)
void gemm_pv_kernel(const __bf16* __restrict__ pb,  // (NB, TP, TP)
                    const __bf16* __restrict__ xb,  // (NB, DF, TP) = v^T rows
                    float* __restrict__ out)        // (NB, DF, TT)
{
    const int  lane = threadIdx.x & 31;
    const int  wave = threadIdx.x >> 5;
    const int  wr   = wave >> 1;
    const int  wc   = wave & 1;
    const int  l15  = lane & 15;
    const long b    = blockIdx.z;
    const long m0   = (long)blockIdx.y * 128;   // query rows (t)
    const long n0   = (long)blockIdx.x * 128;   // feature cols (df)

    v8f acc[2][4] = {};
    gemm_core(pb + (b * TP + m0) * (long)TP, TP,
              xb + (b * DF + n0) * (long)TP, TP,
              TP / 32, acc);

    // transposed store into (NB, DF, TT); guard padded query rows
    #pragma unroll
    for (int mt = 0; mt < 2; ++mt)
        #pragma unroll
        for (int nt = 0; nt < 4; ++nt)
            #pragma unroll
            for (int r = 0; r < 8; ++r) {
                long m = m0 + wr * 32 + mt * 16 + r + ((lane & 16) ? 8 : 0);
                long n = n0 + wc * 64 + nt * 16 + l15;
                if (m < TT)
                    out[(b * DF + n) * (long)TT + m] = acc[mt][nt][r];
            }
}

// ---------------------------------------------------------------------------

extern "C" void kernel_launch(void* const* d_in, const int* in_sizes, int n_in,
                              void* d_out, int out_size, void* d_ws, size_t ws_size,
                              hipStream_t stream)
{
    const float* x = (const float*)d_in[0];   // (4, 64, 64, 2000) f32
    const float* e = (const float*)d_in[1];   // (4, 64, 64)       f32
    float* out = (float*)d_out;               // (4, 64, 64, 2000) f32

    char* ws = (char*)d_ws;
    size_t off = 0;
    __bf16* qb = (__bf16*)(ws + off); off += (size_t)NB * TP * DF * 2;  //  64 MB
    __bf16* xb = (__bf16*)(ws + off); off += (size_t)NB * DF * TP * 2;  //  64 MB
    __bf16* vb = (__bf16*)(ws + off); off += (size_t)NB * DF * TP * 2;  //  64 MB
    float*  sc = (float*) (ws + off); off += (size_t)NB * TP * TP * 4;  //  64 MB
    __bf16* pb = (__bf16*)(ws + off); off += (size_t)NB * TP * TP * 2;  //  32 MB
    (void)ws_size; (void)in_sizes; (void)n_in; (void)out_size;          // 288 MB

    {   // 1) q = (e + PE)/64  (bf16, padded)
        long n = (long)NB * TP * DF;
        prep_q_kernel<<<dim3((unsigned)(n / 256)), dim3(256), 0, stream>>>(e, qb);
    }
    {   // 2) x -> bf16 in both layouts, zero padding
        long n = (long)NB * DF * TP;
        prep_kv_kernel<<<dim3((unsigned)(n / 256)), dim3(256), 0, stream>>>(x, xb, vb);
    }
    // 3) scores = q @ k (scale pre-folded)
    gemm_qk_kernel<<<dim3(TP / 128, TP / 128, NB), dim3(256), 0, stream>>>(qb, vb, sc);
    // 4) P = softmax(scores) masked, bf16
    softmax_kernel<<<dim3(TP, NB), dim3(256), 0, stream>>>(sc, pb);
    // 5) out = (P @ v)^T
    gemm_pv_kernel<<<dim3(DF / 128, TP / 128, NB), dim3(256), 0, stream>>>(pb, xb, out);
}
